// MelSpectrogramONNX_84593675862402
// MI455X (gfx1250) — compile-verified, matrix operand order checked by reading, not run on hardware
//
#include <hip/hip_runtime.h>

typedef _Float16 h8  __attribute__((ext_vector_type(8)));
typedef _Float16 h16 __attribute__((ext_vector_type(16)));
typedef float    f8  __attribute__((ext_vector_type(8)));

namespace {
constexpr int SEG        = 32000;
constexpr int NFFT       = 1024;
constexpr int HOP        = 160;
constexpr int NMELS      = 160;
constexpr int NFREQ      = 513;
constexpr int NFRAMES    = 201;
constexpr int BATCH      = 256;
constexpr int MTILE      = 64;                  // frames per workgroup (4 WMMA M-tiles)
constexpr int TOT_FRAMES = BATCH * NFRAMES;     // 51456 = 804 * 64 exactly
constexpr int NWG        = TOT_FRAMES / MTILE;  // 804 workgroups, zero waste
constexpr int FREQ_TILES = 33;                  // ceil(513/16)
constexpr int FREQ_PAD   = FREQ_TILES * 16;     // 528
constexpr int K2_PAD     = 544;                 // ceil(513/32)*32  (GEMM2 K)
constexpr int LDA        = NFFT + 8;            // 1032 halves per M-sub-tile segment
constexpr int LDA4       = 4 * LDA + 8;         // 4136 halves per row group:
                                                //   fwA[row][mt][k]; all 4 mt segments
                                                //   within DS 16-bit immediate range
constexpr int LDP        = K2_PAD + 8;          // 552 halves
}

// Load a 16-half WMMA fragment (A row-major, or B as B^T row-major) for the
// 16x16x32 f16 shape.  Per-lane K halves are {o..o+7, o+16..o+23}, o = 8*(lane>=16).
__device__ __forceinline__ h16 load_frag(const _Float16* p) {
    h8 lo = *(const h8*)(p);
    h8 hi = *(const h8*)(p + 16);
    return __builtin_shufflevector(lo, hi, 0, 1, 2, 3, 4, 5, 6, 7,
                                           8, 9, 10, 11, 12, 13, 14, 15);
}

__device__ __forceinline__ f8 wmma16(h16 a, h16 b, f8 c) {
    return __builtin_amdgcn_wmma_f32_16x16x32_f16(
        /*neg_a=*/false, a, /*neg_b=*/false, b,
        /*c_mod=*/(short)0, c, /*reuse_a=*/false, /*reuse_b=*/false);
}

// ---------------------------------------------------------------------------
// Pass 0: convert constants fp32 -> f16 with zero padding.
//   dre/dim : [FREQ_PAD][NFFT]   (rows >= 513 zero)
//   mfb     : [NMELS][K2_PAD]    (cols >= 513 zero)
// ---------------------------------------------------------------------------
__global__ void melspec_preconvert(const float* __restrict__ dft_real,
                                   const float* __restrict__ dft_imag,
                                   const float* __restrict__ mel_fb,
                                   _Float16* __restrict__ dre,
                                   _Float16* __restrict__ dim,
                                   _Float16* __restrict__ mfb) {
    int idx = blockIdx.x * blockDim.x + threadIdx.x;
    if (idx < FREQ_PAD * NFFT) {
        int f = idx / NFFT;
        float re = 0.0f, im = 0.0f;
        if (f < NFREQ) { re = dft_real[idx]; im = dft_imag[idx]; }
        dre[idx] = (_Float16)re;
        dim[idx] = (_Float16)im;
    }
    if (idx < NMELS * K2_PAD) {
        int m = idx / K2_PAD;
        int k = idx - m * K2_PAD;
        float v = (k < NFREQ) ? mel_fb[m * NFREQ + k] : 0.0f;
        mfb[idx] = (_Float16)v;
    }
}

// ---------------------------------------------------------------------------
// Main fused kernel: one workgroup = 64 consecutive frames of the flattened
// (batch, frame) axis.  fwA uses an mt-interleaved layout so all four A
// sub-tile fragments load from ONE base register with immediate DS offsets
// (<= ~8 KB): no in-loop VALU writes to WMMA source registers, hence no
// WMMA->VALU hazard NOPs.
// ---------------------------------------------------------------------------
__global__ __launch_bounds__(256) void melspec_main(
    const float* __restrict__ wav,      // [256][32000]
    const float* __restrict__ window,   // [1024]
    const _Float16* __restrict__ dre,   // [528][1024]
    const _Float16* __restrict__ dim,   // [528][1024]
    const _Float16* __restrict__ mfb,   // [160][544]
    float* __restrict__ out)            // [256][160][201]
{
    __shared__ __align__(16) _Float16 fwA[16 * LDA4];    // windowed frames (~132 KB)
    __shared__ __align__(16) _Float16 pw[MTILE * LDP];   // power spectrum (~71 KB)

    const int fbase = blockIdx.x * MTILE;   // first global frame of this tile
    const int tid   = threadIdx.x;
    const int lane  = tid & 31;
    const int wv    = tid >> 5;             // wave id 0..7
    const int row   = lane & 15;            // fragment row/col index
    const int ko    = (lane >> 4) * 8;      // per-lane K offset (0 or 8)

    // --- Stage windowed frames into LDS (reflect padding inline) ---
    // frame r (row = r&15, mt = r>>4) lives at fwA[(r&15)*LDA4 + (r>>4)*LDA].
    for (int e = tid; e < MTILE * NFFT; e += 256) {
        int r = e >> 10;                    // frame row within tile
        int n = e & (NFFT - 1);             // sample within frame
        unsigned g = fbase + r;             // global frame index
        unsigned bb = g / NFRAMES;          // batch
        int t = (int)(g - bb * NFRAMES);    // frame within batch
        int s = t * HOP + n - (NFFT / 2);
        if (s < 0) s = -s;
        if (s >= SEG) s = 2 * SEG - 2 - s;
        fwA[(r & 15) * LDA4 + (r >> 4) * LDA + n] =
            (_Float16)(wav[(size_t)bb * SEG + s] * window[n]);
    }
    // Zero the GEMM2 K-pad columns of the power tile: [FREQ_PAD, K2_PAD)
    for (int e = tid; e < MTILE * (K2_PAD - FREQ_PAD); e += 256) {
        int r = e >> 4;
        int c = e & 15;
        pw[r * LDP + FREQ_PAD + c] = (_Float16)0.0f;
    }
    __syncthreads();

    // Single A-fragment base pointer; mt sub-tiles are immediate offsets.
    const _Float16* const ap = fwA + row * LDA4 + ko;
    const int mbase = (lane >> 4) * 8;      // D layout: M = v + 8*(lane/16)

    // --- GEMM1: frames x DFT; one B (re,im) pair feeds 4 M sub-tiles. ---
#pragma unroll 1
    for (int ftile = wv; ftile < FREQ_TILES; ftile += 8) {
        f8 acc_re[4] = {{}, {}, {}, {}};
        f8 acc_im[4] = {{}, {}, {}, {}};
        const unsigned boff = (unsigned)(ftile * 16 + row) * NFFT + ko;
#pragma unroll 2
        for (int kb = 0; kb < NFFT; kb += 32) {
            h16 br = load_frag(dre + boff + kb);
            h16 bi = load_frag(dim + boff + kb);
            h16 a0 = load_frag(ap + kb);
            h16 a1 = load_frag(ap + LDA + kb);
            h16 a2 = load_frag(ap + 2 * LDA + kb);
            h16 a3 = load_frag(ap + 3 * LDA + kb);
            acc_re[0] = wmma16(a0, br, acc_re[0]);
            acc_im[0] = wmma16(a0, bi, acc_im[0]);
            acc_re[1] = wmma16(a1, br, acc_re[1]);
            acc_im[1] = wmma16(a1, bi, acc_im[1]);
            acc_re[2] = wmma16(a2, br, acc_re[2]);
            acc_im[2] = wmma16(a2, bi, acc_im[2]);
            acc_re[3] = wmma16(a3, br, acc_re[3]);
            acc_im[3] = wmma16(a3, bi, acc_im[3]);
        }
        // power -> LDS, stored [frame][freq] (== B^T row-major for GEMM2)
#pragma unroll
        for (int mt = 0; mt < 4; ++mt) {
#pragma unroll
            for (int v = 0; v < 8; ++v) {
                float p = acc_re[mt][v] * acc_re[mt][v] + acc_im[mt][v] * acc_im[mt][v];
                pw[(mt * 16 + mbase + v) * LDP + ftile * 16 + row] = (_Float16)p;
            }
        }
    }
    __syncthreads();

    // --- GEMM2: mel filterbank x power.  40 output tiles (10 mel x 4 frame). ---
#pragma unroll 1
    for (int idx = wv; idx < (NMELS / 16) * 4; idx += 8) {
        const int mt = idx >> 2;            // mel tile 0..9
        const int nt = idx & 3;             // frame sub-tile 0..3
        f8 acc = {};
        const unsigned aoff = (unsigned)(mt * 16 + row) * K2_PAD + ko;
        const _Float16* bp  = pw + (nt * 16 + row) * LDP + ko;  // B^T row = frame
#pragma unroll 4
        for (int kb = 0; kb < K2_PAD; kb += 32) {
            h16 a  = load_frag(mfb + aoff + kb);
            h16 bb = load_frag(bp + kb);
            acc = wmma16(a, bb, acc);
        }
        // D tile: N (lane%16) = frame, M (v + 8*(lane/16)) = mel
        unsigned g  = fbase + nt * 16 + row;               // global frame (< 51456)
        unsigned bb = g / NFRAMES;                         // batch
        int t = (int)(g - bb * NFRAMES);                   // frame within batch
        const int mel0 = mt * 16 + mbase;
        float* op = out + ((size_t)bb * NMELS + mel0) * NFRAMES + t;
#pragma unroll
        for (int v = 0; v < 8; ++v) {
            op[(size_t)v * NFRAMES] = acc[v];
        }
    }
}

// ---------------------------------------------------------------------------
extern "C" void kernel_launch(void* const* d_in, const int* in_sizes, int n_in,
                              void* d_out, int out_size, void* d_ws, size_t ws_size,
                              hipStream_t stream) {
    const float* wav      = (const float*)d_in[0];   // [256*32000]
    const float* window   = (const float*)d_in[1];   // [1024]
    const float* dft_real = (const float*)d_in[2];   // [513*1024]
    const float* dft_imag = (const float*)d_in[3];   // [513*1024]
    const float* mel_fb   = (const float*)d_in[4];   // [160*513]
    float* out = (float*)d_out;                      // [256*160*201]

    // Workspace carve-up (f16 constants, ~2.3 MB)
    _Float16* dre = (_Float16*)d_ws;
    _Float16* dim = dre + (size_t)FREQ_PAD * NFFT;
    _Float16* mfb = dim + (size_t)FREQ_PAD * NFFT;

    {
        int n = FREQ_PAD * NFFT;                     // 540672 (covers mel_fb too)
        int blocks = (n + 255) / 256;
        melspec_preconvert<<<blocks, 256, 0, stream>>>(dft_real, dft_imag, mel_fb,
                                                       dre, dim, mfb);
    }
    {
        melspec_main<<<NWG, 256, 0, stream>>>(wav, window, dre, dim, mfb, out);
    }
}
